// BeAttention_68899865362562
// MI455X (gfx1250) — matrix operand order repaired and checked by reading, not compiled, not run on hardware
//
#include <hip/hip_runtime.h>
#include <hip/hip_bf16.h>

// ---------------------------------------------------------------------------
// Fused attention for MI455X (gfx1250): bf16 WMMA, f32 accumulate,
// double-buffered operand loads so WMMAs overlap memory latency.
//   B=4, S=2048, D=1024.
// Workspace (~70 MB):
//   [ 0,16MB)   xb   : x as bf16            [B*S, D]
//   [16,22MB)   Wqb/Wkb/Wvb bf16            [D, D] each
//   [22,38MB)   Qb   : (xWq^T+bq)/sqrt(D)   [B*S, D]   bf16
//   [38,54MB)   Kb   :  xWk^T+bk            [B*S, D]   bf16
//   [54,70MB)   Vt   : (xWv^T+bv) transposed per batch [B, D, S] bf16
// ---------------------------------------------------------------------------

typedef __bf16 bf16;
typedef __attribute__((ext_vector_type(16))) __bf16 v16bf;
typedef __attribute__((ext_vector_type(8)))  __bf16 v8bf;
typedef __attribute__((ext_vector_type(8)))  float  v8f;
typedef __attribute__((ext_vector_type(4)))  int    v4i;

union ABf { v16bf v; v8bf h[2]; };

#define B_ 4
#define S_ 2048
#define D_ 1024

static __device__ __forceinline__ v8f wmma_bf16(v16bf a, v16bf b, v8f c) {
  return __builtin_amdgcn_wmma_f32_16x16x32_bf16(false, a, false, b, (short)0, c,
                                                 false, false);
}

#if defined(__HIP_DEVICE_COMPILE__) && \
    __has_builtin(__builtin_amdgcn_global_load_async_to_lds_b128)
#define HAVE_ASYNC_LDS 1
typedef __attribute__((address_space(1))) v4i* gv4i_p;
typedef __attribute__((address_space(3))) v4i* lv4i_p;
#endif

// ---------------------------------------------------------------------------
__global__ void to_bf16_kernel(const float* __restrict__ src,
                               bf16* __restrict__ dst, size_t n) {
  size_t i = (size_t)blockIdx.x * blockDim.x + threadIdx.x;
  const size_t stride = (size_t)gridDim.x * blockDim.x;
  for (; i < n; i += stride) dst[i] = (bf16)src[i];
}

// ---------------------------------------------------------------------------
// Projection GEMM: Out = (X @ W^T + bias) * scale
// 128 threads = 4 waves; WG tile 128x64; wave tile 32x64 (2 M x 4 N WMMA tiles).
// K-loop double-buffered: loads for k+32 issue before WMMAs for k.
// ---------------------------------------------------------------------------
__global__ __launch_bounds__(128) void qkv_gemm_kernel(
    const bf16* __restrict__ X, const bf16* __restrict__ W,
    const float* __restrict__ bias, bf16* __restrict__ Out,
    float scale, int storeT) {
  const int tid  = threadIdx.x;
  const int wave = tid >> 5, lane = tid & 31;
  const int hi = lane >> 4, lo = lane & 15;
  const int rowbase = blockIdx.x * 128 + wave * 32;
  const int colblk  = blockIdx.y * 64;

  const bf16* __restrict__ arow0 = X + (size_t)(rowbase + lo) * D_ + hi * 16;
  const bf16* __restrict__ arow1 = arow0 + 16 * D_;
  const bf16* __restrict__ bp[4];
#pragma unroll
  for (int nt = 0; nt < 4; ++nt)
    bp[nt] = W + (size_t)(colblk + nt * 16 + lo) * D_ + hi * 16;

  v8f acc[8];
#pragma unroll
  for (int i = 0; i < 8; ++i) acc[i] = (v8f){0, 0, 0, 0, 0, 0, 0, 0};

  ABf a[2][2], bm[2][4];
  auto load_k = [&](int buf, int k) {
    a[buf][0].h[0] = *(const v8bf*)(arow0 + k);
    a[buf][0].h[1] = *(const v8bf*)(arow0 + k + 8);
    a[buf][1].h[0] = *(const v8bf*)(arow1 + k);
    a[buf][1].h[1] = *(const v8bf*)(arow1 + k + 8);
#pragma unroll
    for (int nt = 0; nt < 4; ++nt) {
      bm[buf][nt].h[0] = *(const v8bf*)(bp[nt] + k);
      bm[buf][nt].h[1] = *(const v8bf*)(bp[nt] + k + 8);
    }
  };

  load_k(0, 0);
  for (int k = 0; k < D_; k += 32) {
    const int cur = (k >> 5) & 1, nxt = cur ^ 1;
    if (k + 32 < D_) load_k(nxt, k + 32);   // prefetch next K-step
#pragma unroll
    for (int mt = 0; mt < 2; ++mt)
#pragma unroll
      for (int nt = 0; nt < 4; ++nt)
        acc[mt * 4 + nt] = wmma_bf16(a[cur][mt].v, bm[cur][nt].v, acc[mt * 4 + nt]);
  }

#pragma unroll
  for (int nt = 0; nt < 4; ++nt) {
    const int col = colblk + nt * 16 + lo;
    const float bv = bias[col];
#pragma unroll
    for (int mt = 0; mt < 2; ++mt)
#pragma unroll
      for (int r = 0; r < 8; ++r) {
        const int row = rowbase + mt * 16 + hi * 8 + r;
        const float v = (acc[mt * 4 + nt][r] + bv) * scale;
        if (storeT) {
          const int bb = row >> 11, ss = row & (S_ - 1);
          Out[((size_t)bb * D_ + col) * S_ + ss] = (bf16)v;
        } else {
          Out[(size_t)row * D_ + col] = (bf16)v;
        }
      }
  }
}

// ---------------------------------------------------------------------------
// Attention: per (batch b, 16-row query tile). 512 threads = 16 waves.
// LDS: Q tile bf16 [16][1032], P bf16 [16][2056], reductions.
// ---------------------------------------------------------------------------
#define QSTR 1032
#define PSTR 2056
#define SM_QOFF   0
#define SM_POFF   33024            // 16*1032*2
#define SM_REDOFF 98816            // + 16*2056*2
#define SMEM_BYTES 100992          // + 16*32*4 + 16*4 + 16*4

__global__ __launch_bounds__(512) void attn_kernel(
    const bf16* __restrict__ Qb, const bf16* __restrict__ Kb,
    const bf16* __restrict__ Vt, float* __restrict__ Out) {
  extern __shared__ char smem[];
  bf16*  qs     = (bf16*)(smem + SM_QOFF);
  bf16*  Ps     = (bf16*)(smem + SM_POFF);
  float* red    = (float*)(smem + SM_REDOFF);  // [16][32]
  float* rowmax = red + 16 * 32;               // [16]
  float* rowinv = rowmax + 16;                 // [16]

  const int tid = threadIdx.x;
  const int b   = blockIdx.y;
  const int q0  = blockIdx.x * 16;

  // ---- Stage Q tile (16 x 1024 bf16 = 32 KB) into LDS ---------------------
  const bf16* __restrict__ Qsrc = Qb + ((size_t)b * S_ + q0) * D_;
#pragma unroll
  for (int i = 0; i < 4; ++i) {
    const int c   = tid + i * 512;       // 2048 chunks of 16 B
    const int row = c >> 7;
    const int cc  = (c & 127) * 8;
    const bf16* gp = Qsrc + (size_t)row * D_ + cc;
    bf16*       lp = qs + row * QSTR + cc;
#ifdef HAVE_ASYNC_LDS
    __builtin_amdgcn_global_load_async_to_lds_b128((gv4i_p)(void*)gp,
                                                   (lv4i_p)(void*)lp, 0, 0);
#else
    *(v8bf*)lp = *(const v8bf*)gp;
#endif
  }
#ifdef HAVE_ASYNC_LDS
#if __has_builtin(__builtin_amdgcn_s_wait_asynccnt)
  __builtin_amdgcn_s_wait_asynccnt(0);
#else
  asm volatile("s_wait_asynccnt 0x0" ::: "memory");
#endif
#endif
  __syncthreads();

  const int wavei = tid >> 5, lane = tid & 31;
  const int hi = lane >> 4, lo = lane & 15;

  // ---- Phase 1: raw scores S = (Q/sqrt(D)) @ K^T -> bf16 in LDS -----------
  const bf16* __restrict__ qrow = qs + lo * QSTR + hi * 16;
  for (int nt = 0; nt < 8; ++nt) {                 // wave covers 128 cols
    const int col0 = wavei * 128 + nt * 16;
    const bf16* __restrict__ kp =
        Kb + ((size_t)b * S_ + col0 + lo) * D_ + hi * 16;
    if (nt + 1 < 8) __builtin_prefetch(kp + 16 * D_, 0, 1);

    v8f acc = (v8f){0, 0, 0, 0, 0, 0, 0, 0};
    ABf a[2], bm[2];
    a[0].h[0]  = *(const v8bf*)(qrow);
    a[0].h[1]  = *(const v8bf*)(qrow + 8);
    bm[0].h[0] = *(const v8bf*)(kp);
    bm[0].h[1] = *(const v8bf*)(kp + 8);
    for (int k = 0; k < D_; k += 32) {
      const int cur = (k >> 5) & 1, nxt = cur ^ 1;
      if (k + 32 < D_) {
        a[nxt].h[0]  = *(const v8bf*)(qrow + k + 32);
        a[nxt].h[1]  = *(const v8bf*)(qrow + k + 40);
        bm[nxt].h[0] = *(const v8bf*)(kp + k + 32);
        bm[nxt].h[1] = *(const v8bf*)(kp + k + 40);
      }
      acc = wmma_bf16(a[cur].v, bm[cur].v, acc);
    }
#pragma unroll
    for (int r = 0; r < 8; ++r)
      Ps[(r + hi * 8) * PSTR + col0 + lo] = (bf16)acc[r];
  }
  __syncthreads();

  // ---- Softmax (f32 math; P left unnormalized bf16; 1/sum deferred) -------
  {
    const int rr = tid >> 5, c0 = tid & 31;   // 32 threads per row
    bf16* prow = Ps + rr * PSTR;
    float m = -3.0e38f;
    for (int j = c0; j < S_; j += 32) m = fmaxf(m, (float)prow[j]);
    red[rr * 32 + c0] = m;
    __syncthreads();
    if (tid < 16) {
      float mm = red[tid * 32];
#pragma unroll
      for (int j = 1; j < 32; ++j) mm = fmaxf(mm, red[tid * 32 + j]);
      rowmax[tid] = mm;
    }
    __syncthreads();
    const float rm = rowmax[rr];
    float s = 0.f;
    for (int j = c0; j < S_; j += 32) {
      const float e = __expf((float)prow[j] - rm);
      prow[j] = (bf16)e;
      s += e;
    }
    red[rr * 32 + c0] = s;
    __syncthreads();
    if (tid < 16) {
      float ss = 0.f;
#pragma unroll
      for (int j = 0; j < 32; ++j) ss += red[tid * 32 + j];
      rowinv[tid] = 1.0f / ss;
    }
    __syncthreads();
  }

  // ---- Phase 2: O = P @ V (transposed V -> contiguous B loads) ------------
  const int d0 = wavei * 64;                   // wave covers 64 output dims
  v8f oacc[4];
#pragma unroll
  for (int nt = 0; nt < 4; ++nt) oacc[nt] = (v8f){0, 0, 0, 0, 0, 0, 0, 0};
  const bf16* __restrict__ prow2 = Ps + lo * PSTR + hi * 16;
  const bf16* __restrict__ vpb[4];
#pragma unroll
  for (int nt = 0; nt < 4; ++nt)
    vpb[nt] = Vt + ((size_t)b * D_ + d0 + nt * 16 + lo) * S_ + hi * 16;

  ABf a2[2], bv2[2][4];
  a2[0].h[0] = *(const v8bf*)(prow2);
  a2[0].h[1] = *(const v8bf*)(prow2 + 8);
#pragma unroll
  for (int nt = 0; nt < 4; ++nt) {
    bv2[0][nt].h[0] = *(const v8bf*)(vpb[nt]);
    bv2[0][nt].h[1] = *(const v8bf*)(vpb[nt] + 8);
  }
  for (int kk = 0; kk < S_; kk += 32) {
    const int cur = (kk >> 5) & 1, nxt = cur ^ 1;
    if (kk + 32 < S_) {
      a2[nxt].h[0] = *(const v8bf*)(prow2 + kk + 32);
      a2[nxt].h[1] = *(const v8bf*)(prow2 + kk + 40);
#pragma unroll
      for (int nt = 0; nt < 4; ++nt) {
        bv2[nxt][nt].h[0] = *(const v8bf*)(vpb[nt] + kk + 32);
        bv2[nxt][nt].h[1] = *(const v8bf*)(vpb[nt] + kk + 40);
      }
    }
#pragma unroll
    for (int nt = 0; nt < 4; ++nt)
      oacc[nt] = wmma_bf16(a2[cur].v, bv2[cur][nt].v, oacc[nt]);
  }
#pragma unroll
  for (int nt = 0; nt < 4; ++nt) {
    const int col = d0 + nt * 16 + lo;
#pragma unroll
    for (int r = 0; r < 8; ++r) {
      const int row = hi * 8 + r;
      Out[((size_t)b * S_ + q0 + row) * D_ + col] = oacc[nt][r] * rowinv[row];
    }
  }
}

// ---------------------------------------------------------------------------
extern "C" void kernel_launch(void* const* d_in, const int* in_sizes, int n_in,
                              void* d_out, int out_size, void* d_ws,
                              size_t ws_size, hipStream_t stream) {
  const float* x  = (const float*)d_in[0];
  const float* Wq = (const float*)d_in[1];
  const float* bq = (const float*)d_in[2];
  const float* Wk = (const float*)d_in[3];
  const float* bk = (const float*)d_in[4];
  const float* Wv = (const float*)d_in[5];
  const float* bv = (const float*)d_in[6];
  float* out = (float*)d_out;

  char* ws = (char*)d_ws;
  const size_t NX = (size_t)B_ * S_ * D_;  // 8M
  const size_t NW = (size_t)D_ * D_;       // 1M
  bf16* xb  = (bf16*)ws;
  bf16* Wqb = (bf16*)(ws + (16u << 20));
  bf16* Wkb = Wqb + NW;
  bf16* Wvb = Wkb + NW;
  bf16* Qb  = (bf16*)(ws + (22u << 20));
  bf16* Kb  = Qb + NX;
  bf16* Vt  = Kb + NX;

  to_bf16_kernel<<<2048, 256, 0, stream>>>(x, xb, NX);
  to_bf16_kernel<<<512, 256, 0, stream>>>(Wq, Wqb, NW);
  to_bf16_kernel<<<512, 256, 0, stream>>>(Wk, Wkb, NW);
  to_bf16_kernel<<<512, 256, 0, stream>>>(Wv, Wvb, NW);

  dim3 ggrid(64, 16);  // 8192/128 row tiles x 1024/64 col tiles
  qkv_gemm_kernel<<<ggrid, 128, 0, stream>>>(xb, Wqb, bq, Qb, 0.03125f, 0);
  qkv_gemm_kernel<<<ggrid, 128, 0, stream>>>(xb, Wkb, bk, Kb, 1.0f, 0);
  qkv_gemm_kernel<<<ggrid, 128, 0, stream>>>(xb, Wvb, bv, Vt, 1.0f, 1);

  attn_kernel<<<dim3(S_ / 16, B_), 512, SMEM_BYTES, stream>>>(Qb, Kb, Vt, out);
}
